// ZNGraphConv_13589276524721
// MI455X (gfx1250) — compile-verified, matrix operand order checked by reading, not run on hardware
//
#include <hip/hip_runtime.h>
#include <hip/hip_bf16.h>

typedef _Float16 v16h __attribute__((ext_vector_type(16)));
typedef _Float16 v8h  __attribute__((ext_vector_type(8)));
typedef float    v8f  __attribute__((ext_vector_type(8)));
typedef float    v4f  __attribute__((ext_vector_type(4)));

#define DIN    256
#define DOUT   256
#define DSUP   128
#define NCOMB  384          // DOUT + DSUP combined output columns
#define KSTEPS 8            // 256 / 32
#define NT_OUT 16           // N tiles going to d_out (256 cols)
#define NT_SUP 8            // N tiles going to w1 scratch (128 cols)

// d_ws layout (bytes):
#define WS_WF16_OFF 0           // f16 weights, 384*256*2 = 196608
#define WS_BIAS_OFF 196608      // f32 combined bias, 384*4 = 1536
#define WS_STAGE_BYTES 198144   // weights + bias, staged to LDS as one blob
#define WS_W1_OFF   262144      // f32 verts_w1 buffer, V*128*4 = 102.4 MB

#define STAGE_CHUNKS (WS_STAGE_BYTES / 16)   // 12384 x b128

// ---------------------------------------------------------------------------
// Pack [w0_w ; w1_w] into one row-major 384x256 f16 matrix + combined bias.
// ---------------------------------------------------------------------------
__global__ __launch_bounds__(256)
void zn_convert_weights(const float* __restrict__ w0w, const float* __restrict__ w0b,
                        const float* __restrict__ w1w, const float* __restrict__ w1b,
                        _Float16* __restrict__ wf16, float* __restrict__ biasf) {
    int i = blockIdx.x * blockDim.x + threadIdx.x;
    if (i < DOUT * DIN)            wf16[i] = (_Float16)w0w[i];
    else if (i < NCOMB * DIN)      wf16[i] = (_Float16)w1w[i - DOUT * DIN];
    if (i < NCOMB)                 biasf[i] = (i < DOUT) ? w0b[i] : w1b[i - DOUT];
}

// ---------------------------------------------------------------------------
// One 16x16 output tile: 16 ds_load_b128 of B + 8 chained WMMAs.
// ---------------------------------------------------------------------------
__device__ __forceinline__
v8f zn_tile_mm(const _Float16* __restrict__ bp, const v16h* __restrict__ a) {
    v16h b[KSTEPS];
#pragma unroll
    for (int ks = 0; ks < KSTEPS; ++ks) {
        v8h b0 = *(const v8h*)(bp + ks * 32);
        v8h b1 = *(const v8h*)(bp + ks * 32 + 16);
        v16h t;
#pragma unroll
        for (int j = 0; j < 8; ++j) { t[j] = b0[j]; t[8 + j] = b1[j]; }
        b[ks] = t;
    }
    v8f c = {0.f, 0.f, 0.f, 0.f, 0.f, 0.f, 0.f, 0.f};
#pragma unroll
    for (int ks = 0; ks < KSTEPS; ++ks)
        c = __builtin_amdgcn_wmma_f32_16x16x32_f16(
                false, a[ks], false, b[ks], (short)0, c, false, false);
    return c;
}

// ---------------------------------------------------------------------------
// Fused dual GEMM: out[v, 0:256] = verts[v] @ w0^T + b0  (d_out)
//                  w1out[v,0:128] = verts[v] @ w1^T + b1 (scratch)
// Per block (8 waves): async-DMA the full 384x256 f16 weight matrix + bias
// into LDS, overlapped with the A-panel f32->f16 conversion loads. One wave
// owns a 16-row M tile; A fragments stay register-resident across the sweep.
// ---------------------------------------------------------------------------
__global__ __launch_bounds__(256)
void zn_gemm(const float* __restrict__ verts,
             const char* __restrict__ wsrc,   // f16 weights + f32 bias, contiguous
             float* __restrict__ out,         // V x 256
             float* __restrict__ w1out,       // V x 128
             int Vn) {
    extern __shared__ char smem[];            // WS_STAGE_BYTES dynamic LDS
    const int tid  = threadIdx.x;
    const int lane = tid & 31;
    const int half = lane >> 4;               // 16-lane half
    const int l16  = lane & 15;
    const int m0   = blockIdx.x * 128 + (tid >> 5) * 16;

    // ---- issue async DMA: weights+bias blob -> LDS (ASYNCcnt-tracked) ----
    for (int idx = tid; idx < STAGE_CHUNKS; idx += 256) {
        unsigned ldsoff = (unsigned)(size_t)(void*)(smem + idx * 16);
        const char* gp  = wsrc + (long)idx * 16;
        asm volatile("global_load_async_to_lds_b128 %0, %1, off"
                     :: "v"(ldsoff), "v"(gp) : "memory");
    }

    // ---- load + convert A panel while the DMA flies ----
    // WMMA 16-bit A layout: lane holds row (m0+l16); per k-step the lane's
    // K chunks are [half*8, half*8+8) and [16+half*8, 16+half*8+8).
    v16h a[KSTEPS];
    {
        int row = m0 + l16;
        if (row >= Vn) row = Vn - 1;          // clamp: keep loads in-bounds
        const float* ap = verts + (long)row * DIN + half * 8;
#pragma unroll
        for (int ks = 0; ks < KSTEPS; ++ks) {
            const float* p = ap + ks * 32;
            v4f f0 = *(const v4f*)(p);
            v4f f1 = *(const v4f*)(p + 4);
            v4f f2 = *(const v4f*)(p + 16);
            v4f f3 = *(const v4f*)(p + 20);
            v16h t;
#pragma unroll
            for (int j = 0; j < 4; ++j) {
                t[j]      = (_Float16)f0[j];
                t[4 + j]  = (_Float16)f1[j];
                t[8 + j]  = (_Float16)f2[j];
                t[12 + j] = (_Float16)f3[j];
            }
            a[ks] = t;
        }
    }

    asm volatile("s_wait_asynccnt 0x0" ::: "memory");
    __syncthreads();
    if (__builtin_amdgcn_readfirstlane(m0) >= Vn) return;  // uniform exit

    const _Float16* bsh    = (const _Float16*)smem;
    const float*    biassh = (const float*)(smem + WS_BIAS_OFF);
    const int rbase = m0 + half * 8;

    // wave-uniform tail decision, forced scalar
    const int fullTile =
        __builtin_amdgcn_readfirstlane((m0 + 16 <= Vn) ? 1 : 0);

    if (fullTile) {
        // ---- hot path: no guards, clean clause stores ----
        for (int nt = 0; nt < NT_OUT; ++nt) {
            const int ncol = nt * 16 + l16;
            v8f c = zn_tile_mm(bsh + ncol * DIN + half * 8, a);
            const float bias = biassh[ncol];
            float* op = out + (long)rbase * DOUT + ncol;
#pragma unroll
            for (int r = 0; r < 8; ++r) op[(long)r * DOUT] = c[r] + bias;
        }
        for (int nt = 0; nt < NT_SUP; ++nt) {
            const int ncol = (NT_OUT + nt) * 16 + l16;       // 256..383
            v8f c = zn_tile_mm(bsh + ncol * DIN + half * 8, a);
            const float bias = biassh[ncol];
            float* op = w1out + (long)rbase * DSUP + (ncol - DOUT);
#pragma unroll
            for (int r = 0; r < 8; ++r) op[(long)r * DSUP] = c[r] + bias;
        }
    } else {
        // ---- ragged last block only ----
        for (int nt = 0; nt < NT_OUT; ++nt) {
            const int ncol = nt * 16 + l16;
            v8f c = zn_tile_mm(bsh + ncol * DIN + half * 8, a);
            const float bias = biassh[ncol];
            float* op = out + (long)rbase * DOUT + ncol;
#pragma unroll
            for (int r = 0; r < 8; ++r)
                if (rbase + r < Vn) op[(long)r * DOUT] = c[r] + bias;
        }
        for (int nt = 0; nt < NT_SUP; ++nt) {
            const int ncol = (NT_OUT + nt) * 16 + l16;
            v8f c = zn_tile_mm(bsh + ncol * DIN + half * 8, a);
            const float bias = biassh[ncol];
            float* op = w1out + (long)rbase * DSUP + (ncol - DOUT);
#pragma unroll
            for (int r = 0; r < 8; ++r)
                if (rbase + r < Vn) op[(long)r * DSUP] = c[r] + bias;
        }
    }
}

// ---------------------------------------------------------------------------
// Undirected edge scatter: for directed edge (dst, src), out[dst, 0:128] +=
// w1[src, 0:128]. One wave per directed edge; each lane owns 4 columns.
// w1 buffer (102 MB) is L2-resident on MI455X (192 MB L2).
// ---------------------------------------------------------------------------
__global__ __launch_bounds__(256)
void zn_scatter(const int* __restrict__ edges,
                const float* __restrict__ w1,
                float* __restrict__ out, int En) {
    const int lane = threadIdx.x & 31;
    const int e = blockIdx.x * 8 + (threadIdx.x >> 5);
    if (e >= 2 * En) return;

    int dst, src;
    if (e < En) { dst = edges[2 * e];            src = edges[2 * e + 1]; }
    else        { int e2 = e - En;
                  dst = edges[2 * e2 + 1];       src = edges[2 * e2]; }

    const v4f val = *(const v4f*)(w1 + (long)src * DSUP + lane * 4);
    float* op = out + (long)dst * DOUT + lane * 4;
    atomicAdd(op + 0, val[0]);
    atomicAdd(op + 1, val[1]);
    atomicAdd(op + 2, val[2]);
    atomicAdd(op + 3, val[3]);
}

// ---------------------------------------------------------------------------
extern "C" void kernel_launch(void* const* d_in, const int* in_sizes, int n_in,
                              void* d_out, int out_size, void* d_ws, size_t ws_size,
                              hipStream_t stream) {
    const float* verts = (const float*)d_in[0];
    const int*   edges = (const int*)  d_in[1];
    const float* w0w   = (const float*)d_in[2];
    const float* w0b   = (const float*)d_in[3];
    const float* w1w   = (const float*)d_in[4];
    const float* w1b   = (const float*)d_in[5];

    const int Vn = in_sizes[0] / DIN;
    const int En = in_sizes[1] / 2;

    float*    out   = (float*)d_out;
    char*     ws    = (char*)d_ws;
    _Float16* wf16  = (_Float16*)(ws + WS_WF16_OFF);
    float*    biasf = (float*)(ws + WS_BIAS_OFF);
    float*    w1buf = (float*)(ws + WS_W1_OFF);

    // allow > default dynamic LDS (198144 B; gfx1250 WGP has 320 KB)
    static bool attrDone = false;
    if (!attrDone) {
        (void)hipFuncSetAttribute((const void*)zn_gemm,
                                  hipFuncAttributeMaxDynamicSharedMemorySize,
                                  WS_STAGE_BYTES);
        attrDone = true;
    }

    // 1) pack weights/bias to f16 (tiny)
    zn_convert_weights<<<(NCOMB * DIN + 255) / 256, 256, 0, stream>>>(
        w0w, w0b, w1w, w1b, wf16, biasf);

    // 2) fused WMMA dual-GEMM (writes all of d_out + w1 scratch)
    zn_gemm<<<(Vn + 127) / 128, 256, WS_STAGE_BYTES, stream>>>(
        verts, ws, out, w1buf, Vn);

    // 3) edge gather + atomic scatter into d_out columns [0,128)
    zn_scatter<<<(2 * En + 7) / 8, 256, 0, stream>>>(edges, w1buf, out, En);
}
